// qMoEModelBatched_6743098655632
// MI455X (gfx1250) — 8-bit matrix operand order repaired and checked by reading, compiled, not run on hardware
//
#include <hip/hip_runtime.h>
#include <cstdint>
#include <cstddef>

typedef __attribute__((ext_vector_type(16))) int   v16i;
typedef __attribute__((ext_vector_type(8)))  float v8f;
typedef int v4i_gcc __attribute__((vector_size(16)));

#define BDIM 256
#define B_TOK 8192
#define IN_DIM 512
#define HID 2048
#define NCLS 1000
#define NCLS_PAD 1024
#define NEXP 4
#define RH 256
#define RH2 128

#define FP8_POS1 ((uint8_t)0x38)
#define FP8_NEG1 ((uint8_t)0xB8)
#define FP8_ZERO ((uint8_t)0x00)

__device__ __forceinline__ uint8_t fp8_sign(float v) {
  return (v > 0.0f) ? FP8_POS1 : ((v < 0.0f) ? FP8_NEG1 : FP8_ZERO);
}

#if __has_builtin(__builtin_amdgcn_global_load_async_to_lds_b128)
#define HAVE_ASYNC_LDS 1
#else
#define HAVE_ASYNC_LDS 0
#endif

#if HAVE_ASYNC_LDS
#define GAS __attribute__((address_space(1)))
#define LAS __attribute__((address_space(3)))

__device__ __forceinline__ void async_b128(const uint8_t* gsrc, uint32_t lds_byte_off) {
  __builtin_amdgcn_global_load_async_to_lds_b128(
      (GAS v4i_gcc*)(uintptr_t)gsrc, (LAS v4i_gcc*)(uintptr_t)lds_byte_off, 0, 0);
}

__device__ __forceinline__ void wait_asynccnt0() {
#if __has_builtin(__builtin_amdgcn_s_wait_asynccnt)
  __builtin_amdgcn_s_wait_asynccnt(0);
#else
  asm volatile("s_wait_asynccnt 0" ::: "memory");
#endif
}
#endif

__global__ __launch_bounds__(BDIM) void zero_f32_kernel(float* __restrict__ p, long n) {
  long stride = (long)gridDim.x * BDIM;
  for (long i = (long)blockIdx.x * BDIM + threadIdx.x; i < n; i += stride) p[i] = 0.0f;
}

__global__ __launch_bounds__(BDIM) void sign_f32_to_fp8_kernel(const float* __restrict__ x,
                                                               uint8_t* __restrict__ y, long n) {
  long stride = (long)gridDim.x * BDIM;
  for (long i = (long)blockIdx.x * BDIM + threadIdx.x; i < n; i += stride)
    y[i] = fp8_sign(x[i]);
}

#define CHUNKS 64
__global__ __launch_bounds__(BDIM) void wsum_partial_kernel(const float* __restrict__ W,
                                                            float* __restrict__ partial,
                                                            long S  ) {
  int blk = blockIdx.x;
  int e = blk / CHUNKS;
  int c = blk % CHUNKS;
  const float* w = W + (size_t)e * S;
  float s = 0.0f;
  long step = (long)CHUNKS * BDIM;
  for (long i = (long)c * BDIM + threadIdx.x; i < S; i += step) s += w[i];
  __shared__ float red[BDIM];
  red[threadIdx.x] = s;
  __syncthreads();
  for (int st = BDIM / 2; st > 0; st >>= 1) {
    if (threadIdx.x < st) red[threadIdx.x] += red[threadIdx.x + st];
    __syncthreads();
  }
  if (threadIdx.x == 0) partial[blk] = red[0];
}

__global__ void wmean_final_kernel(const float* __restrict__ partial, float* __restrict__ mean,
                                   long S) {
  int e = threadIdx.x;
  if (e >= NEXP) return;
  float s = 0.0f;
  for (int c = 0; c < CHUNKS; ++c) s += partial[e * CHUNKS + c];
  mean[e] = s / (float)S;
}

__global__ __launch_bounds__(BDIM) void ternarize_kernel(const float* __restrict__ W,
                                                         const float* __restrict__ mean,
                                                         uint8_t* __restrict__ T,
                                                         long rowsIn, long rowsOut, long K) {
  long total = (long)NEXP * rowsOut * K;
  long stride = (long)gridDim.x * BDIM;
  for (long idx = (long)blockIdx.x * BDIM + threadIdx.x; idx < total; idx += stride) {
    long e = idx / (rowsOut * K);
    long rem = idx - e * rowsOut * K;
    long r = rem / K;
    long k = rem - r * K;
    uint8_t v = FP8_ZERO;
    if (r < rowsIn) v = fp8_sign(W[(e * rowsIn + r) * K + k] - mean[e]);
    T[idx] = v;
  }
}

#define RTOK 32
__global__ __launch_bounds__(BDIM) void router_fc_kernel(const float* __restrict__ X,
                                                         const float* __restrict__ W,
                                                         const float* __restrict__ bias,
                                                         float* __restrict__ Y,
                                                         int K, int Nout) {
  extern __shared__ float xr[];
  int b0 = blockIdx.x * RTOK;
  for (int idx = threadIdx.x; idx < RTOK * K; idx += BDIM) {
    int t = idx / K, i = idx - t * K;
    xr[idx] = X[(size_t)(b0 + t) * K + i];
  }
  __syncthreads();
  int o = threadIdx.x;
  if (o < Nout) {
    const float* w = W + (size_t)o * K;
    float acc[RTOK];
#pragma unroll
    for (int t = 0; t < RTOK; ++t) acc[t] = bias[o];
    for (int i = 0; i < K; ++i) {
      float wv = w[i];
#pragma unroll
      for (int t = 0; t < RTOK; ++t) acc[t] = fmaf(xr[t * K + i], wv, acc[t]);
    }
#pragma unroll
    for (int t = 0; t < RTOK; ++t)
      Y[(size_t)(b0 + t) * Nout + o] = fmaxf(acc[t], 0.0f);
  }
}

__global__ __launch_bounds__(BDIM) void router_top2_kernel(const float* __restrict__ H,
                                                           const float* __restrict__ Wr3,
                                                           const float* __restrict__ br3,
                                                           float* __restrict__ gate,
                                                           int* __restrict__ topi) {
  int b = blockIdx.x * BDIM + threadIdx.x;
  if (b >= B_TOK) return;
  const float* h = H + (size_t)b * RH2;
  float s[NEXP];
#pragma unroll
  for (int e = 0; e < NEXP; ++e) {
    float a = br3[e];
    const float* w = Wr3 + (size_t)e * RH2;
    for (int i = 0; i < RH2; ++i) a = fmaf(h[i], w[i], a);
    s[e] = a;
  }
  float m = s[0];
#pragma unroll
  for (int e = 1; e < NEXP; ++e) m = fmaxf(m, s[e]);
  float p[NEXP];
#pragma unroll
  for (int e = 0; e < NEXP; ++e) p[e] = __expf(s[e] - m);
  int i0 = 0;
#pragma unroll
  for (int e = 1; e < NEXP; ++e) if (p[e] > p[i0]) i0 = e;
  int i1 = (i0 == 0) ? 1 : 0;
#pragma unroll
  for (int e = 0; e < NEXP; ++e) if (e != i0 && e != i1 && p[e] > p[i1]) i1 = e;
  float denom = p[i0] + p[i1];
  gate[(size_t)b * 2 + 0] = p[i0] / denom;
  gate[(size_t)b * 2 + 1] = p[i1] / denom;
  topi[(size_t)b * 2 + 0] = i0;
  topi[(size_t)b * 2 + 1] = i1;
}

#define BM 128
#define BN 128
#define BK 128
#define TILE_DW (BM * (BK / 4))

__global__ __launch_bounds__(BDIM) void gemm_fp8_kernel(const uint8_t* __restrict__ A,
                                                        const uint8_t* __restrict__ Bw,
                                                        float* __restrict__ C,
                                                        int M, int N, int K) {
  const int t = threadIdx.x;
  const int wave = t >> 5;
  const int lane = t & 31;
  const int hi = lane >> 4;
  const int lo = lane & 15;
  const int wm = wave >> 1;
  const int wn = wave & 1;
  const int m0 = blockIdx.y * BM;
  const int n0 = blockIdx.x * BN;

  v8f acc[2][4];
#pragma unroll
  for (int mi = 0; mi < 2; ++mi)
#pragma unroll
    for (int ni = 0; ni < 4; ++ni) acc[mi][ni] = {};

  const int ldr = t >> 3;
  const int seg = t & 7;

#if HAVE_ASYNC_LDS
  __shared__ __align__(16) uint32_t lA[2][TILE_DW];
  __shared__ __align__(16) uint32_t lB[2][TILE_DW];
  const uint32_t aBase = (uint32_t)(uintptr_t)&lA[0][0];
  const uint32_t bBase = (uint32_t)(uintptr_t)&lB[0][0];
  const int ksteps = K / BK;

#pragma unroll
  for (int p = 0; p < 4; ++p) {
    int r = ldr + p * 32;
    async_b128(A + (size_t)(m0 + r) * K + seg * 16, aBase + r * BK + seg * 16);
    async_b128(Bw + (size_t)(n0 + r) * K + seg * 16, bBase + r * BK + seg * 16);
  }
  wait_asynccnt0();
  __syncthreads();

  for (int ks = 0; ks < ksteps; ++ks) {
    const int buf = ks & 1;
    if (ks + 1 < ksteps) {
      const int nb = (ks + 1) & 1;
      const int kb = (ks + 1) * BK;
#pragma unroll
      for (int p = 0; p < 4; ++p) {
        int r = ldr + p * 32;
        async_b128(A + (size_t)(m0 + r) * K + kb + seg * 16,
                   aBase + (uint32_t)nb * (TILE_DW * 4) + r * BK + seg * 16);
        async_b128(Bw + (size_t)(n0 + r) * K + kb + seg * 16,
                   bBase + (uint32_t)nb * (TILE_DW * 4) + r * BK + seg * 16);
      }
    }
    const uint32_t* tA = &lA[buf][0];
    const uint32_t* tB = &lB[buf][0];
#else
  __shared__ __align__(16) uint32_t lA1[TILE_DW];
  __shared__ __align__(16) uint32_t lB1[TILE_DW];
  for (int kb = 0; kb < K; kb += BK) {
#pragma unroll
    for (int p = 0; p < 4; ++p) {
      int r = ldr + p * 32;
      ((int4*)lA1)[r * 8 + seg] =
          *(const int4*)(A + (size_t)(m0 + r) * K + kb + seg * 16);
      ((int4*)lB1)[r * 8 + seg] =
          *(const int4*)(Bw + (size_t)(n0 + r) * K + kb + seg * 16);
    }
    __syncthreads();
    const uint32_t* tA = &lA1[0];
    const uint32_t* tB = &lB1[0];
#endif

    v16i af[2];
#pragma unroll
    for (int mi = 0; mi < 2; ++mi) {
      const uint32_t* base = tA + (wm * 32 + mi * 16 + lo) * 32 + hi * 4;
      v16i v;
#pragma unroll
      for (int h = 0; h < 4; ++h) {
        uint4 q = *(const uint4*)(base + h * 8);
        v[h * 4 + 0] = (int)q.x;
        v[h * 4 + 1] = (int)q.y;
        v[h * 4 + 2] = (int)q.z;
        v[h * 4 + 3] = (int)q.w;
      }
      af[mi] = v;
    }

#pragma unroll
    for (int ni = 0; ni < 4; ++ni) {
      const uint32_t* base = tB + (wn * 64 + ni * 16 + lo) * 32 + hi * 4;
      v16i bf;
#pragma unroll
      for (int h = 0; h < 4; ++h) {
        uint4 q = *(const uint4*)(base + h * 8);
        bf[h * 4 + 0] = (int)q.x;
        bf[h * 4 + 1] = (int)q.y;
        bf[h * 4 + 2] = (int)q.z;
        bf[h * 4 + 3] = (int)q.w;
      }
      acc[0][ni] = __builtin_amdgcn_wmma_f32_16x16x128_fp8_fp8(
          af[0], bf, (short)0, acc[0][ni], false, false);
      acc[1][ni] = __builtin_amdgcn_wmma_f32_16x16x128_fp8_fp8(
          af[1], bf, (short)0, acc[1][ni], false, false);
    }

#if HAVE_ASYNC_LDS
    wait_asynccnt0();
    __syncthreads();
  }
#else
    __syncthreads();
  }
#endif

#pragma unroll
  for (int mi = 0; mi < 2; ++mi)
#pragma unroll
    for (int ni = 0; ni < 4; ++ni) {
      int rowb = m0 + wm * 32 + mi * 16 + hi * 8;
      int col = n0 + wn * 64 + ni * 16 + lo;
#pragma unroll
      for (int r = 0; r < 8; ++r)
        C[(size_t)(rowb + r) * N + col] = acc[mi][ni][r];
    }
}

__global__ __launch_bounds__(BDIM) void ln_relu_sign_kernel(const float* __restrict__ H,
                                                            const float* __restrict__ g,
                                                            const float* __restrict__ beta,
                                                            uint8_t* __restrict__ A, int N) {
  int b = blockIdx.x;
  const float* h = H + (size_t)b * N;
  float s = 0.0f, s2 = 0.0f;
  for (int i = threadIdx.x; i < N; i += BDIM) {
    float v = h[i];
    s += v;
    s2 += v * v;
  }
  __shared__ float r1[BDIM], r2[BDIM];
  r1[threadIdx.x] = s;
  r2[threadIdx.x] = s2;
  __syncthreads();
  for (int st = BDIM / 2; st > 0; st >>= 1) {
    if (threadIdx.x < st) {
      r1[threadIdx.x] += r1[threadIdx.x + st];
      r2[threadIdx.x] += r2[threadIdx.x + st];
    }
    __syncthreads();
  }
  __shared__ float mu_s, rstd_s;
  if (threadIdx.x == 0) {
    float mu = r1[0] / (float)N;
    float var = r2[0] / (float)N - mu * mu;
    mu_s = mu;
    rstd_s = rsqrtf(var + 1e-5f);
  }
  __syncthreads();
  float mu = mu_s, rstd = rstd_s;
  for (int i = threadIdx.x; i < N; i += BDIM) {
    float v = (h[i] - mu) * rstd * g[i] + beta[i];
    A[(size_t)b * N + i] = (v > 0.0f) ? FP8_POS1 : FP8_ZERO;
  }
}

__global__ __launch_bounds__(BDIM) void combine_kernel(const float* __restrict__ EO,
                                                       const float* __restrict__ gate,
                                                       const int* __restrict__ topi,
                                                       float* __restrict__ out, int e) {
  int b = blockIdx.x;
  float gsum = 0.0f;
  if (topi[(size_t)b * 2 + 0] == e) gsum += gate[(size_t)b * 2 + 0];
  if (topi[(size_t)b * 2 + 1] == e) gsum += gate[(size_t)b * 2 + 1];
  if (gsum == 0.0f) return;
  for (int c = threadIdx.x; c < NCLS; c += BDIM)
    out[(size_t)b * NCLS + c] += gsum * EO[(size_t)b * NCLS_PAD + c];
}

static inline void* ws_take(char*& p, size_t bytes) {
  void* r = (void*)p;
  p += (bytes + 255) & ~(size_t)255;
  return r;
}

extern "C" void kernel_launch(void* const* d_in, const int* in_sizes, int n_in,
                              void* d_out, int out_size, void* d_ws, size_t ws_size,
                              hipStream_t stream) {
  (void)in_sizes; (void)n_in; (void)out_size; (void)ws_size;
  const float* x   = (const float*)d_in[0];
  const float* Wr1 = (const float*)d_in[1];
  const float* br1 = (const float*)d_in[2];
  const float* Wr2 = (const float*)d_in[3];
  const float* br2 = (const float*)d_in[4];
  const float* Wr3 = (const float*)d_in[5];
  const float* br3 = (const float*)d_in[6];
  const float* W1  = (const float*)d_in[7];
  const float* g1  = (const float*)d_in[8];
  const float* b1  = (const float*)d_in[9];
  const float* W2  = (const float*)d_in[10];
  const float* g2  = (const float*)d_in[11];
  const float* b2  = (const float*)d_in[12];
  const float* W3  = (const float*)d_in[13];
  float* out = (float*)d_out;

  char* p = (char*)d_ws;
  uint8_t* xs   = (uint8_t*)ws_take(p, (size_t)B_TOK * IN_DIM);
  uint8_t* tw1  = (uint8_t*)ws_take(p, (size_t)NEXP * HID * IN_DIM);
  uint8_t* tw2  = (uint8_t*)ws_take(p, (size_t)NEXP * HID * HID);
  uint8_t* tw3  = (uint8_t*)ws_take(p, (size_t)NEXP * NCLS_PAD * HID);
  uint8_t* act  = (uint8_t*)ws_take(p, (size_t)B_TOK * HID);
  float*   Hbuf = (float*)  ws_take(p, (size_t)B_TOK * HID * sizeof(float));
  float*   rh1  = (float*)  ws_take(p, (size_t)B_TOK * RH * sizeof(float));
  float*   rh2  = (float*)  ws_take(p, (size_t)B_TOK * RH2 * sizeof(float));
  float*   gate = (float*)  ws_take(p, (size_t)B_TOK * 2 * sizeof(float));
  int*     topi = (int*)    ws_take(p, (size_t)B_TOK * 2 * sizeof(int));
  float*   part = (float*)  ws_take(p, (size_t)NEXP * CHUNKS * sizeof(float));
  float*   mean = (float*)  ws_take(p, 16 * sizeof(float));

  zero_f32_kernel<<<4096, BDIM, 0, stream>>>(out, (long)B_TOK * NCLS);

  sign_f32_to_fp8_kernel<<<4096, BDIM, 0, stream>>>(x, xs, (long)B_TOK * IN_DIM);

  const long S1 = (long)HID * IN_DIM, S2 = (long)HID * HID, S3 = (long)NCLS * HID;
  wsum_partial_kernel<<<NEXP * CHUNKS, BDIM, 0, stream>>>(W1, part, S1);
  wmean_final_kernel<<<1, 32, 0, stream>>>(part, mean + 0, S1);
  ternarize_kernel<<<4096, BDIM, 0, stream>>>(W1, mean + 0, tw1, HID, HID, IN_DIM);
  wsum_partial_kernel<<<NEXP * CHUNKS, BDIM, 0, stream>>>(W2, part, S2);
  wmean_final_kernel<<<1, 32, 0, stream>>>(part, mean + 4, S2);
  ternarize_kernel<<<8192, BDIM, 0, stream>>>(W2, mean + 4, tw2, HID, HID, HID);
  wsum_partial_kernel<<<NEXP * CHUNKS, BDIM, 0, stream>>>(W3, part, S3);
  wmean_final_kernel<<<1, 32, 0, stream>>>(part, mean + 8, S3);
  ternarize_kernel<<<8192, BDIM, 0, stream>>>(W3, mean + 8, tw3, NCLS, NCLS_PAD, HID);

  router_fc_kernel<<<B_TOK / RTOK, BDIM, RTOK * IN_DIM * sizeof(float), stream>>>(
      x, Wr1, br1, rh1, IN_DIM, RH);
  router_fc_kernel<<<B_TOK / RTOK, BDIM, RTOK * RH * sizeof(float), stream>>>(
      rh1, Wr2, br2, rh2, RH, RH2);
  router_top2_kernel<<<B_TOK / BDIM, BDIM, 0, stream>>>(rh2, Wr3, br3, gate, topi);

  for (int e = 0; e < NEXP; ++e) {
    gemm_fp8_kernel<<<dim3(HID / BN, B_TOK / BM), BDIM, 0, stream>>>(
        xs, tw1 + (size_t)e * HID * IN_DIM, Hbuf, B_TOK, HID, IN_DIM);
    ln_relu_sign_kernel<<<B_TOK, BDIM, 0, stream>>>(
        Hbuf, g1 + (size_t)e * HID, b1 + (size_t)e * HID, act, HID);
    gemm_fp8_kernel<<<dim3(HID / BN, B_TOK / BM), BDIM, 0, stream>>>(
        act, tw2 + (size_t)e * HID * HID, Hbuf, B_TOK, HID, HID);
    ln_relu_sign_kernel<<<B_TOK, BDIM, 0, stream>>>(
        Hbuf, g2 + (size_t)e * HID, b2 + (size_t)e * HID, act, HID);
    gemm_fp8_kernel<<<dim3(NCLS_PAD / BN, B_TOK / BM), BDIM, 0, stream>>>(
        act, tw3 + (size_t)e * NCLS_PAD * HID, Hbuf, B_TOK, NCLS_PAD, HID);
    combine_kernel<<<B_TOK, BDIM, 0, stream>>>(Hbuf, gate, topi, out, e);
  }
}
